// SparseBlock_25658134626630
// MI455X (gfx1250) — matrix operand order repaired and checked by reading, not compiled, not run on hardware
//
#include <hip/hip_runtime.h>

#define DIM   96
#define HID   384
#define MB    128          // points per block tile
#define NTHR  128          // 4 waves, 32 rows each
#define LDA   104          // bf16 element stride (208B = 13*16 -> aligned, conflict-free)
#define LDC   100          // f32 element stride for conv-out tile

// conv-kernel LDS map (bytes)
#define BUFSZ   46592      // one double-buffer slot: As(26624) + BsT(19968)
#define XS_OFF  93184      // after both buffers
#define SSQ_OFF 119808     // [4][HID] f32
#define SMEM_SZ 125952

typedef __attribute__((ext_vector_type(16))) __bf16 v16bf;
typedef __attribute__((ext_vector_type(8)))  float  v8f;

union FragU { v16bf v; uint4 q[2]; };

// round-to-nearest-even f32 -> bf16 without relying on fptrunc lowering
__device__ inline __bf16 f2bf(float x) {
    union { float f; unsigned int u; } v; v.f = x;
    unsigned int r = v.u + 0x7FFFu + ((v.u >> 16) & 1u);
    union { unsigned short s; __bf16 b; } o; o.s = (unsigned short)(r >> 16);
    return o.b;
}

// A fragment: 16x32 bf16, rows striped lanes 0..15, K split per ISA layout
__device__ inline v16bf load_fragA(const __bf16* As, int lane, int mrow, int k0) {
    const __bf16* p = As + (mrow + (lane & 15)) * LDA + k0;
    int hi = (lane >> 4) & 1;              // lanes 16..31 hold K+8 / K+24
    FragU f;
    f.q[0] = *(const uint4*)(p + hi * 8);
    f.q[1] = *(const uint4*)(p + 16 + hi * 8);
    return f.v;
}

// B fragment from N-major LDS tile BsT[N][K]: lane holds col n0+(lane&15),
// K = k0 + (hi?8:0) + j and k0 + 16 + (hi?8:0) + j  -> two 16-byte chunks
__device__ inline v16bf load_fragB(const __bf16* BsT, int lane, int n0, int k0) {
    const __bf16* p = BsT + (n0 + (lane & 15)) * LDA + k0 + ((lane >> 4) & 1) * 8;
    FragU f;
    f.q[0] = *(const uint4*)(p);
    f.q[1] = *(const uint4*)(p + 16);
    return f.v;
}

__device__ inline v8f wmma_bf16(v16bf a, v16bf b, v8f c) {
    return __builtin_amdgcn_wmma_f32_16x16x32_bf16(false, a, false, b, (short)0, c, false, false);
}

// CDNA5 async DMA: global -> LDS, 16 bytes per active lane (ASYNCcnt-tracked)
__device__ inline void async_copy_b128(unsigned int lds_off, const void* gaddr) {
    asm volatile("global_load_async_to_lds_b128 %0, %1, off"
                 :: "v"(lds_off), "v"(gaddr) : "memory");
}
__device__ inline void wait_asynccnt0() {
    asm volatile("s_wait_asynccnt 0x0" ::: "memory");
}
__device__ inline unsigned int lds_off32(const void* p) {
    // generic->LDS mapping truncates to addr[31:0] (ISA 10.2 aperture rules)
    return (unsigned int)(unsigned long long)p;
}

// ---------------- prep: f32 -> bf16 conversions -------------------------
__global__ void prep_kernel(const float* __restrict__ feats,
                            const float* __restrict__ Wdw,
                            const float* __restrict__ W1,
                            __bf16* __restrict__ feats_bf,
                            __bf16* __restrict__ WdwT,
                            __bf16* __restrict__ W1b,
                            long nFeat, int Koff) {
    long stride = (long)gridDim.x * blockDim.x;
    long i0 = (long)blockIdx.x * blockDim.x + threadIdx.x;
    for (long i = i0; i < nFeat; i += stride) feats_bf[i] = f2bf(feats[i]);
    long nW = (long)Koff * DIM * DIM;
    for (long i = i0; i < nW; i += stride) {           // transpose to [k][co][ci]
        int k = (int)(i / (DIM * DIM));
        int r = (int)(i % (DIM * DIM));
        int co = r / DIM, ci = r % DIM;
        WdwT[i] = f2bf(Wdw[(size_t)k * DIM * DIM + (size_t)ci * DIM + co]);
    }
    for (long i = i0; i < (long)HID * DIM; i += stride) W1b[i] = f2bf(W1[i]); // already [j][c]
}

// -------- fused sparse-conv + LayerNorm + pwconv1 + ReLU + GRN partials --
__global__ __launch_bounds__(NTHR)
void conv_ln_pw1_kernel(const int* __restrict__ nbr,
                        const float* __restrict__ b_dw,
                        const float* __restrict__ ln_w,
                        const float* __restrict__ ln_b,
                        const float* __restrict__ b1,
                        const __bf16* __restrict__ feats_bf,
                        const __bf16* __restrict__ WdwT,
                        const __bf16* __restrict__ W1b,
                        __bf16* __restrict__ h_bf,
                        float* __restrict__ part,
                        int Ntot, int Koff) {
    __shared__ __attribute__((aligned(16))) unsigned char smem[SMEM_SZ];
    float*  Cs   = (float*) (smem);            // overlays buffers after conv
    __bf16* W1T  = (__bf16*)(smem);            // overlays Cs during pw1
    __bf16* Xs   = (__bf16*)(smem + XS_OFF);   // 128 x LDA bf16
    float*  ssqw = (float*) (smem + SSQ_OFF);  // [4][HID]

    const int tid  = threadIdx.x;
    const int lane = tid & 31;
    const int wave = tid >> 5;
    const int mrow = wave * 32;                // 32 rows per wave (2 A-fragments)
    const int block0 = blockIdx.x * MB;
    const int rows_valid = min(MB, Ntot - block0);

    const v8f zero8 = {0.f,0.f,0.f,0.f,0.f,0.f,0.f,0.f};
    v8f acc0[6], acc1[6];
#pragma unroll
    for (int n = 0; n < 6; ++n) { acc0[n] = zero8; acc1[n] = zero8; }

    int g[12];  // prefetched neighbor indices for the NEXT stage (1536 = 128*12)
    auto fetch_idx = [&](int k) {
#pragma unroll
        for (int i = 0; i < 12; ++i) {
            int row = (tid + NTHR * i) / 12;
            g[i] = (row < rows_valid) ? nbr[(size_t)k * Ntot + block0 + row] : -1;
        }
    };
    auto stage = [&](int k) {                         // async global->LDS, no waits
        unsigned char* base = smem + (k & 1) * BUFSZ;
        __bf16* Asb = (__bf16*)(base);
        __bf16* Bsb = (__bf16*)(base + 26624);
#pragma unroll
        for (int i = 0; i < 12; ++i) {                // gather 128x96 bf16 tile
            int c = tid + NTHR * i;
            int row = c / 12, sg = c % 12;
            __bf16* dst = Asb + row * LDA + sg * 8;
            if (g[i] >= 0)
                async_copy_b128(lds_off32(dst), feats_bf + (size_t)g[i] * DIM + sg * 8);
            else
                *(uint4*)dst = make_uint4(0u, 0u, 0u, 0u);
        }
        const __bf16* wsrc = WdwT + (size_t)k * (DIM * DIM);
#pragma unroll
        for (int i = 0; i < 9; ++i) {                 // weight tile [co][ci] (1152 = 128*9)
            int c = tid + NTHR * i;
            int co = c / 12, sg = c % 12;
            async_copy_b128(lds_off32(Bsb + co * LDA + sg * 8), wsrc + co * DIM + sg * 8);
        }
    };
    auto compute = [&](int k) {
        const __bf16* Asb = (const __bf16*)(smem + (k & 1) * BUFSZ);
        const __bf16* Bsb = (const __bf16*)(smem + (k & 1) * BUFSZ + 26624);
#pragma unroll
        for (int kk = 0; kk < 3; ++kk) {
            v16bf a0 = load_fragA(Asb, lane, mrow,      kk * 32);
            v16bf a1 = load_fragA(Asb, lane, mrow + 16, kk * 32);
#pragma unroll
            for (int n = 0; n < 6; ++n) {             // each B fragment feeds 2 wmmas
                v16bf b = load_fragB(Bsb, lane, n * 16, kk * 32);
                acc0[n] = wmma_bf16(a0, b, acc0[n]);
                acc1[n] = wmma_bf16(a1, b, acc1[n]);
            }
        }
    };

    // ---- software-pipelined sparse conv: stage(k+1) overlaps compute(k) ----
    fetch_idx(0);
    stage(0);
    if (Koff > 1) fetch_idx(1);
    for (int k = 0; k < Koff; ++k) {
        wait_asynccnt0();                 // own stage(k) DMA done
        __syncthreads();                  // everyone's stage(k) visible; buf[(k+1)&1] free
        if (k + 1 < Koff) {
            stage(k + 1);                 // async, overlapped with compute(k)
            if (k + 2 < Koff) fetch_idx(k + 2);
        }
        compute(k);
    }

    // -------- +b_dw, spill to LDS (f32), LayerNorm -> Xs (bf16) ----------
    __syncthreads();
#pragma unroll
    for (int n = 0; n < 6; ++n) {
        int col = n * 16 + (lane & 15);
        float bd = b_dw[col];
        int rb = ((lane >> 4) & 1) * 8;
#pragma unroll
        for (int r = 0; r < 8; ++r) {
            Cs[(mrow +      rb + r) * LDC + col] = acc0[n][r] + bd;
            Cs[(mrow + 16 + rb + r) * LDC + col] = acc1[n][r] + bd;
        }
    }
    __syncthreads();
    {
        const float* xr = Cs + tid * LDC;             // one row per thread (128 threads)
        float mu = 0.f;
        for (int c = 0; c < DIM; ++c) mu += xr[c];
        mu *= (1.f / DIM);
        float var = 0.f;
        for (int c = 0; c < DIM; ++c) { float d = xr[c] - mu; var += d * d; }
        var *= (1.f / DIM);
        float rs = rsqrtf(var + 1e-6f);
        __bf16* xo = Xs + tid * LDA;
        for (int c = 0; c < DIM; ++c)
            xo[c] = f2bf((xr[c] - mu) * rs * ln_w[c] + ln_b[c]);
    }

    // -------- pwconv1 (96 -> 384 in 4 chunks of 96) + ReLU + GRN partials
    for (int q = 0; q < 4; ++q) {
        __syncthreads();                              // Cs dead / prev W1T reads done
        const __bf16* wsrc = W1b + (size_t)q * DIM * DIM;
#pragma unroll
        for (int i = 0; i < 9; ++i) {
            int c = tid + NTHR * i;
            int jl = c / 12, sg = c % 12;
            *(uint4*)(W1T + jl * LDA + sg * 8) = *(const uint4*)(wsrc + jl * DIM + sg * 8);
        }
        __syncthreads();
        v8f h0[6], h1[6];
#pragma unroll
        for (int n = 0; n < 6; ++n) { h0[n] = zero8; h1[n] = zero8; }
#pragma unroll
        for (int kk = 0; kk < 3; ++kk) {
            v16bf a0 = load_fragA(Xs, lane, mrow,      kk * 32);
            v16bf a1 = load_fragA(Xs, lane, mrow + 16, kk * 32);
#pragma unroll
            for (int n = 0; n < 6; ++n) {
                v16bf b = load_fragB(W1T, lane, n * 16, kk * 32);
                h0[n] = wmma_bf16(a0, b, h0[n]);
                h1[n] = wmma_bf16(a1, b, h1[n]);
            }
        }
#pragma unroll
        for (int n = 0; n < 6; ++n) {
            int jcol = q * DIM + n * 16 + (lane & 15);
            float bb = b1[jcol];
            int rb = ((lane >> 4) & 1) * 8;
            float ss = 0.f;
#pragma unroll
            for (int r = 0; r < 8; ++r) {
                float v0 = fmaxf(h0[n][r] + bb, 0.f);
                float v1 = fmaxf(h1[n][r] + bb, 0.f);
                int row0 = mrow + rb + r;
                int row1 = mrow + 16 + rb + r;
                if (row0 < rows_valid) {
                    ss += v0 * v0;
                    h_bf[(size_t)(block0 + row0) * HID + jcol] = f2bf(v0);
                }
                if (row1 < rows_valid) {
                    ss += v1 * v1;
                    h_bf[(size_t)(block0 + row1) * HID + jcol] = f2bf(v1);
                }
            }
            ss += __shfl_xor(ss, 16);                // combine row halves (deterministic)
            if (lane < 16) ssqw[wave * HID + jcol] = ss;
        }
    }
    __syncthreads();
    for (int c = tid; c < HID; c += NTHR) {          // deterministic block partial
        float tot = 0.f;
#pragma unroll
        for (int w = 0; w < 4; ++w) tot += ssqw[w * HID + c];
        part[(size_t)blockIdx.x * HID + c] = tot;
    }
}

// -------- GRN finalize: Nx, fold scale into W2, fold beta into bias -----
__global__ void grn_prep_kernel(const float* __restrict__ part, int nblocks,
                                const float* __restrict__ gamma,
                                const float* __restrict__ beta,
                                const float* __restrict__ W2,
                                const float* __restrict__ b2,
                                __bf16* __restrict__ W2s,
                                float* __restrict__ bias2) {
    __shared__ float gx[HID];
    __shared__ float meansh;
    int c = threadIdx.x;                             // 384 threads
    float s = 0.f;
    for (int b = 0; b < nblocks; ++b) s += part[(size_t)b * HID + c];
    float g = sqrtf(s);
    gx[c] = g;
    __syncthreads();
    if (c == 0) {
        float m = 0.f;
        for (int i = 0; i < HID; ++i) m += gx[i];
        meansh = m / HID;
    }
    __syncthreads();
    float nx = g / (meansh + 1e-6f);
    float sc = 1.f + gamma[c] * nx;                  // h' = h*(1+g*Nx) + beta
    for (int j = 0; j < DIM; ++j)
        W2s[(size_t)j * HID + c] = f2bf(W2[(size_t)j * HID + c] * sc);
    if (c < DIM) {
        float bb = b2[c];
        for (int cc = 0; cc < HID; ++cc) bb += beta[cc] * W2[(size_t)c * HID + cc];
        bias2[c] = bb;
    }
}

// -------------- pwconv2 (384 -> 96) + bias + residual -------------------
__global__ __launch_bounds__(NTHR)
void pw2_kernel(const float* __restrict__ feats,
                const __bf16* __restrict__ h_bf,
                const __bf16* __restrict__ W2s,
                const float* __restrict__ bias2,
                float* __restrict__ out, int Ntot) {
    __shared__ __attribute__((aligned(16))) unsigned char smem[46592];
    __bf16* As = (__bf16*)(smem);            // 128 x LDA
    __bf16* Bs = (__bf16*)(smem + 26624);    // 96  x LDA

    const int tid  = threadIdx.x;
    const int lane = tid & 31;
    const int wave = tid >> 5;
    const int mrow = wave * 32;
    const int block0 = blockIdx.x * MB;
    const int rows_valid = min(MB, Ntot - block0);

    const v8f zero8 = {0.f,0.f,0.f,0.f,0.f,0.f,0.f,0.f};
    v8f acc0[6], acc1[6];
#pragma unroll
    for (int n = 0; n < 6; ++n) { acc0[n] = zero8; acc1[n] = zero8; }

    for (int kc = 0; kc < 4; ++kc) {                 // K = 384 in chunks of 96
        __syncthreads();
        for (int t = tid; t < 1536 + 1152; t += NTHR) {
            if (t < 1536) {
                int row = t / 12, seg = t % 12;
                uint4 val = make_uint4(0u, 0u, 0u, 0u);
                if (row < rows_valid)
                    val = *(const uint4*)(h_bf + (size_t)(block0 + row) * HID + kc * DIM + seg * 8);
                *(uint4*)(As + row * LDA + seg * 8) = val;
            } else {
                int s = t - 1536;
                int j = s / 12, seg = s % 12;
                *(uint4*)(Bs + j * LDA + seg * 8) =
                    *(const uint4*)(W2s + (size_t)j * HID + kc * DIM + seg * 8);
            }
        }
        __syncthreads();
#pragma unroll
        for (int kk = 0; kk < 3; ++kk) {
            v16bf a0 = load_fragA(As, lane, mrow,      kk * 32);
            v16bf a1 = load_fragA(As, lane, mrow + 16, kk * 32);
#pragma unroll
            for (int n = 0; n < 6; ++n) {
                v16bf b = load_fragB(Bs, lane, n * 16, kk * 32);
                acc0[n] = wmma_bf16(a0, b, acc0[n]);
                acc1[n] = wmma_bf16(a1, b, acc1[n]);
            }
        }
    }
#pragma unroll
    for (int n = 0; n < 6; ++n) {
        int col = n * 16 + (lane & 15);
        float bz = bias2[col];
        int rb = ((lane >> 4) & 1) * 8;
#pragma unroll
        for (int r = 0; r < 8; ++r) {
            int row0 = mrow + rb + r;
            int row1 = mrow + 16 + rb + r;
            if (row0 < rows_valid) {
                size_t gi = (size_t)(block0 + row0) * DIM + col;
                out[gi] = acc0[n][r] + bz + feats[gi];
            }
            if (row1 < rows_valid) {
                size_t gi = (size_t)(block0 + row1) * DIM + col;
                out[gi] = acc1[n][r] + bz + feats[gi];
            }
        }
    }
}

// ------------------------------ launch ----------------------------------
extern "C" void kernel_launch(void* const* d_in, const int* in_sizes, int n_in,
                              void* d_out, int out_size, void* d_ws, size_t ws_size,
                              hipStream_t stream) {
    const float* feats = (const float*)d_in[0];
    const int*   nbr   = (const int*)  d_in[1];
    const float* Wdw   = (const float*)d_in[2];
    const float* b_dw  = (const float*)d_in[3];
    const float* ln_w  = (const float*)d_in[4];
    const float* ln_b  = (const float*)d_in[5];
    const float* W1    = (const float*)d_in[6];
    const float* b1    = (const float*)d_in[7];
    const float* gamma = (const float*)d_in[8];
    const float* beta  = (const float*)d_in[9];
    const float* W2    = (const float*)d_in[10];
    const float* b2    = (const float*)d_in[11];
    float* out = (float*)d_out;

    const int Ntot = in_sizes[0] / DIM;
    const int Koff = in_sizes[2] / (DIM * DIM);
    const int nblocks = (Ntot + MB - 1) / MB;

    char* ws = (char*)d_ws;
    size_t off = 0;
    auto alloc = [&](size_t bytes) {
        char* p = ws + off;
        off = (off + bytes + 255) & ~(size_t)255;
        return p;
    };
    __bf16* feats_bf = (__bf16*)alloc((size_t)Ntot * DIM * 2);
    __bf16* WdwT     = (__bf16*)alloc((size_t)Koff * DIM * DIM * 2);
    __bf16* W1b      = (__bf16*)alloc((size_t)HID * DIM * 2);
    __bf16* h_bf     = (__bf16*)alloc((size_t)Ntot * HID * 2);
    float*  part     = (float*) alloc((size_t)nblocks * HID * 4);
    __bf16* W2s      = (__bf16*)alloc((size_t)HID * DIM * 2);
    float*  bias2    = (float*) alloc((size_t)DIM * 4);
    if (off > ws_size) return;                       // workspace too small

    prep_kernel<<<2048, 256, 0, stream>>>(feats, Wdw, W1, feats_bf, WdwT, W1b,
                                          (long)Ntot * DIM, Koff);
    conv_ln_pw1_kernel<<<nblocks, NTHR, 0, stream>>>(nbr, b_dw, ln_w, ln_b, b1,
                                                     feats_bf, WdwT, W1b,
                                                     h_bf, part, Ntot, Koff);
    grn_prep_kernel<<<1, HID, 0, stream>>>(part, nblocks, gamma, beta, W2, b2,
                                           W2s, bias2);
    pw2_kernel<<<nblocks, NTHR, 0, stream>>>(feats, h_bf, W2s, bias2, out, Ntot);
}
